// CAM_Module_163208757845
// MI455X (gfx1250) — matrix operand order repaired and checked by reading, not compiled, not run on hardware
//
#include <hip/hip_runtime.h>

#define BB 16
#define CC 512
#define NNE 4096
#define KC 32          // K chunk per stage
#define LDP 40         // padded LDS row length (elements); 80 B, 16 B aligned
#define BUFSZ (128 * LDP)

typedef __bf16 bf16;
typedef bf16 v16bf __attribute__((ext_vector_type(16)));
typedef float v8f  __attribute__((ext_vector_type(8)));

union Frag { v16bf v; uint4 u[2]; };

// ---------------------------------------------------------------------------
// Kernel 0: f32 -> bf16 cast; emit Qb (C x N row-major) and Qt (N x C row-major)
// via an LDS-tiled transpose so both GEMM stages get K-contiguous operands.
// ---------------------------------------------------------------------------
__global__ __launch_bounds__(256) void k_cast_transpose(
    const float* __restrict__ x, bf16* __restrict__ qb, bf16* __restrict__ qt) {
  int bid = blockIdx.x;
  int nt = bid & 127;          // N/32 = 128 tiles
  int ct = (bid >> 7) & 15;    // C/32 = 16 tiles
  int b  = bid >> 11;
  int n0 = nt * 32, c0 = ct * 32;
  __shared__ bf16 tile[32][33];
  const float* xb = x + (size_t)b * CC * NNE;
  bf16* qbb = qb + (size_t)b * CC * NNE;
  bf16* qtb = qt + (size_t)b * NNE * CC;
  int tx = threadIdx.x, ty = threadIdx.y;
#pragma unroll
  for (int i = 0; i < 4; ++i) {
    int cl = ty + i * 8;
    float v = xb[(size_t)(c0 + cl) * NNE + n0 + tx];
    bf16 h = (bf16)v;
    qbb[(size_t)(c0 + cl) * NNE + n0 + tx] = h;   // coalesced along n
    tile[cl][tx] = h;
  }
  __syncthreads();
#pragma unroll
  for (int i = 0; i < 4; ++i) {
    int nl = ty + i * 8;
    qtb[(size_t)(n0 + nl) * CC + c0 + tx] = tile[tx][nl];  // coalesced along c
  }
}

// ---------------------------------------------------------------------------
// Async stage: copy a 128-row x 32-K bf16 slice (row stride `stride`) into an
// LDS buffer using the CDNA5 async-to-LDS path (tracked by ASYNCcnt).
// 512 x 16B chunks, 2 per thread; each lane copies 16 B global -> LDS.
// ---------------------------------------------------------------------------
__device__ inline void stage_async(const bf16* __restrict__ src, int stride,
                                   int kk, bf16* dstBuf, int tid) {
#pragma unroll
  for (int j = 0; j < 2; ++j) {
    int chunk = tid + 256 * j;
    int row  = chunk >> 2;
    int quad = (chunk & 3) * 8;                 // K offset in elements (16 B)
    const bf16* g = src + (size_t)row * stride + kk + quad;
    unsigned l = (unsigned)(uintptr_t)(dstBuf + row * LDP + quad);
    asm volatile("global_load_async_to_lds_b128 %0, %1, off"
                 :: "v"(l), "v"(g) : "memory");
  }
}

__device__ inline void wait_async0() {
  asm volatile("s_wait_asynccnt 0x0" ::: "memory");
}

// ---------------------------------------------------------------------------
// Block GEMM core: 256 threads compute a 128x128 f32 block of A*B^T-style
// product where both operands are row-major with K contiguous.
// Wave w -> (wr = w>>1, wc = w&1): rows wr*32..+31 (2 tiles), cols wc*64..+63
// (4 tiles) => 8 WMMA accumulators per wave. Double-buffered LDS staging.
// ---------------------------------------------------------------------------
template <int KTOT>
__device__ inline void gemm_block_core(const bf16* __restrict__ Abase,
                                       const bf16* __restrict__ Bbase,
                                       int stride, int tid,
                                       bf16* As, bf16* Bs, v8f acc[2][4]) {
  stage_async(Abase, stride, 0, As, tid);
  stage_async(Bbase, stride, 0, Bs, tid);
  wait_async0();
  __syncthreads();

  int wave = tid >> 5, lane = tid & 31;
  int wr = wave >> 1, wc = wave & 1;
  int half = lane >> 4, l15 = lane & 15;
  int buf = 0;
  for (int kk = 0; kk < KTOT; kk += KC) {
    const bf16* Ac = As + buf * BUFSZ;
    const bf16* Bc = Bs + buf * BUFSZ;
    bool more = (kk + KC) < KTOT;
    if (more) {
      stage_async(Abase, stride, kk + KC, As + (buf ^ 1) * BUFSZ, tid);
      stage_async(Bbase, stride, kk + KC, Bs + (buf ^ 1) * BUFSZ, tid);
    }
    Frag a[2], bb[4];
#pragma unroll
    for (int i = 0; i < 2; ++i) {
      const bf16* pa = Ac + (wr * 32 + i * 16 + l15) * LDP + half * 8;
      a[i].u[0] = *(const uint4*)pa;
      a[i].u[1] = *(const uint4*)(pa + 16);
    }
#pragma unroll
    for (int i = 0; i < 4; ++i) {
      const bf16* pb = Bc + (wc * 64 + i * 16 + l15) * LDP + half * 8;
      bb[i].u[0] = *(const uint4*)pb;
      bb[i].u[1] = *(const uint4*)(pb + 16);
    }
#pragma unroll
    for (int i = 0; i < 2; ++i)
#pragma unroll
      for (int jj = 0; jj < 4; ++jj)
        acc[i][jj] = __builtin_amdgcn_wmma_f32_16x16x32_bf16(
            false, a[i].v, false, bb[jj].v, (short)0, acc[i][jj], false, false);
    if (more) wait_async0();
    __syncthreads();
    buf ^= 1;
  }
}

// ---------------------------------------------------------------------------
// Kernel 1: E = Q * Q^T per batch (512x512, K = 4096), 128x128 blocks.
// ---------------------------------------------------------------------------
__global__ __launch_bounds__(256) void k_gram(const bf16* __restrict__ qb,
                                              float* __restrict__ energy) {
  __shared__ bf16 As[2 * BUFSZ];
  __shared__ bf16 Bs[2 * BUFSZ];
  int bid = blockIdx.x;
  int b  = bid >> 4;           // 16 blocks per batch (4x4 of 128x128)
  int t  = bid & 15;
  int cB = (t >> 2) * 128;
  int dB = (t & 3) * 128;
  const bf16* q = qb + (size_t)b * CC * NNE;
  v8f acc[2][4] = {};
  gemm_block_core<NNE>(q + (size_t)cB * NNE, q + (size_t)dB * NNE, NNE,
                       threadIdx.x, As, Bs, acc);
  int wave = threadIdx.x >> 5, lane = threadIdx.x & 31;
  int wr = wave >> 1, wc = wave & 1;
  int half = lane >> 4, l15 = lane & 15;
  float* e = energy + (size_t)b * CC * CC;
  int col0 = dB + wc * 64 + l15;
#pragma unroll
  for (int i = 0; i < 2; ++i)
#pragma unroll
    for (int r = 0; r < 8; ++r) {
      int m = cB + wr * 32 + i * 16 + half * 8 + r;
#pragma unroll
      for (int jj = 0; jj < 4; ++jj)
        e[(size_t)m * CC + col0 + jj * 16] = acc[i][jj][r];
    }
}

// ---------------------------------------------------------------------------
// Kernel 2: attention = softmax(rowmax(E) - E) via shift invariance:
// att[c,d] = exp(min_c - e[c,d]) / sum_d exp(min_c - e[c,d]).
// One wave per 512-wide row, wave32 shuffle reductions, bf16 output.
// ---------------------------------------------------------------------------
__global__ __launch_bounds__(256) void k_softmax(const float* __restrict__ energy,
                                                 bf16* __restrict__ att) {
  int row  = blockIdx.x * 8 + (threadIdx.x >> 5);   // B*C rows
  int lane = threadIdx.x & 31;
  const float4* rp = (const float4*)(energy + (size_t)row * CC) + lane * 4;
  union { float4 v4[4]; float f[16]; } rc;
  rc.v4[0] = rp[0]; rc.v4[1] = rp[1]; rc.v4[2] = rp[2]; rc.v4[3] = rp[3];
  float mn = rc.f[0];
#pragma unroll
  for (int i = 1; i < 16; ++i) mn = fminf(mn, rc.f[i]);
  for (int m = 16; m; m >>= 1) mn = fminf(mn, __shfl_xor(mn, m, 32));
  float s = 0.f;
#pragma unroll
  for (int i = 0; i < 16; ++i) {
    float w = __expf(mn - rc.f[i]);
    rc.f[i] = w;
    s += w;
  }
  for (int m = 16; m; m >>= 1) s += __shfl_xor(s, m, 32);
  float inv = 1.0f / s;
  union { bf16 h[16]; uint4 u[2]; } o;
#pragma unroll
  for (int i = 0; i < 16; ++i) o.h[i] = (bf16)(rc.f[i] * inv);
  uint4* op = (uint4*)(att + (size_t)row * CC + lane * 16);
  op[0] = o.u[0]; op[1] = o.u[1];
}

// ---------------------------------------------------------------------------
// Kernel 3: out = gamma * (A @ Q) + x.  M=512(c), N=4096(n), K=512(d).
// A rows from attention (K=d contiguous), B rows from Qt (K=d contiguous).
// Same 128x128 block core; fused f32 epilogue.
// ---------------------------------------------------------------------------
__global__ __launch_bounds__(256) void k_out(const bf16* __restrict__ att,
                                             const bf16* __restrict__ qt,
                                             const float* __restrict__ x,
                                             const float* __restrict__ gamma,
                                             float* __restrict__ out) {
  __shared__ bf16 As[2 * BUFSZ];
  __shared__ bf16 Bs[2 * BUFSZ];
  int bid = blockIdx.x;
  int b  = bid >> 7;           // 128 blocks per batch (4 x 32 of 128x128)
  int t  = bid & 127;
  int cB = (t >> 5) * 128;
  int nB = (t & 31) * 128;
  const bf16* arow = att + (size_t)b * CC * CC + (size_t)cB * CC;
  const bf16* brow = qt + (size_t)b * NNE * CC + (size_t)nB * CC;
  v8f acc[2][4] = {};
  gemm_block_core<CC>(arow, brow, CC, threadIdx.x, As, Bs, acc);
  float g = gamma[0];
  const float* xb = x + (size_t)b * CC * NNE;
  float* ob = out + (size_t)b * CC * NNE;
  int wave = threadIdx.x >> 5, lane = threadIdx.x & 31;
  int wr = wave >> 1, wc = wave & 1;
  int half = lane >> 4, l15 = lane & 15;
  int col0 = nB + wc * 64 + l15;
#pragma unroll
  for (int i = 0; i < 2; ++i)
#pragma unroll
    for (int r = 0; r < 8; ++r) {
      int m = cB + wr * 32 + i * 16 + half * 8 + r;
#pragma unroll
      for (int jj = 0; jj < 4; ++jj) {
        size_t idx = (size_t)m * NNE + col0 + jj * 16;
        ob[idx] = g * acc[i][jj][r] + xb[idx];
      }
    }
}

extern "C" void kernel_launch(void* const* d_in, const int* in_sizes, int n_in,
                              void* d_out, int out_size, void* d_ws, size_t ws_size,
                              hipStream_t stream) {
  const float* x     = (const float*)d_in[0];
  const float* gamma = (const float*)d_in[1];
  float* out = (float*)d_out;

  char* ws = (char*)d_ws;
  const size_t qbytes = (size_t)BB * CC * NNE * sizeof(bf16);   // 64 MiB
  const size_t ebytes = (size_t)BB * CC * CC * sizeof(float);   // 16 MiB
  bf16*  qb     = (bf16*)(ws);
  bf16*  qt     = (bf16*)(ws + qbytes);
  float* energy = (float*)(ws + 2 * qbytes);
  bf16*  att    = (bf16*)(ws + 2 * qbytes + ebytes);

  // 1) f32 -> bf16 (row-major + transposed copies)
  k_cast_transpose<<<dim3(BB * (CC / 32) * (NNE / 32)), dim3(32, 8), 0, stream>>>(x, qb, qt);
  // 2) E = Q Q^T   (16 batches x 16 blocks of 128x128)
  k_gram<<<dim3(BB * 16), dim3(256), 0, stream>>>(qb, energy);
  // 3) row softmax of (max - E)
  k_softmax<<<dim3(BB * CC / 8), dim3(256), 0, stream>>>(energy, att);
  // 4) out = gamma * (A Q) + x   (16 batches x 128 blocks of 128x128)
  k_out<<<dim3(BB * 2048 / 16), dim3(256), 0, stream>>>(att, qt, x, gamma, out);
}
